// Attention_Align_Block_65395172049655
// MI455X (gfx1250) — matrix operand order repaired and checked by reading, not compiled, not run on hardware
//
#include <hip/hip_runtime.h>
#include <hip/hip_bf16.h>

typedef __attribute__((ext_vector_type(16))) _Float16 v16h;
typedef __attribute__((ext_vector_type(8)))  _Float16 v8h;
typedef __attribute__((ext_vector_type(8)))  float    v8f;

constexpr int B = 8, C = 2, F = 256, T = 4096, P = 64, D = 32;
constexpr int IN_DIM = C * F / 4;        // 128
constexpr int BT = B * T;                // 32768 rows
constexpr long OUT_ELEMS = (long)B * C * F * T;  // 16777216

static __device__ inline v16h join16(v8h lo, v8h hi) {
  v16h r;
#pragma unroll
  for (int i = 0; i < 8; ++i) { r[i] = lo[i]; r[i + 8] = hi[i]; }
  return r;
}

// ---------------------------------------------------------------------------
// Kernel 0: cast Wq/Wk (64x128 f32) to f16
// ---------------------------------------------------------------------------
__global__ void cast_w_kernel(const float* __restrict__ wq, const float* __restrict__ wk,
                              _Float16* __restrict__ wq16, _Float16* __restrict__ wk16) {
  int i = blockIdx.x * 256 + threadIdx.x;       // grid 32 * 256 == 8192 == 64*128
  wq16[i] = (_Float16)wq[i];
  wk16[i] = (_Float16)wk[i];
}

// ---------------------------------------------------------------------------
// Kernel 1: maxpool(4) over freq + transpose (B,C,F,T) -> (B*T, 128) f16
// grid (T/64, IN_DIM/32, 2*B), block 256
// ---------------------------------------------------------------------------
__global__ __launch_bounds__(256) void pool_kernel(const float* __restrict__ xm,
                                                   const float* __restrict__ xf,
                                                   _Float16* __restrict__ pm,
                                                   _Float16* __restrict__ pf) {
  __shared__ _Float16 lds[32][66];              // 66-half row stride -> 33-bank step
  int bz = blockIdx.z;
  const float* x;
  _Float16* o;
  int b;
  if (bz < B) { x = xm; o = pm; b = bz; } else { x = xf; o = pf; b = bz - B; }
  int t0 = blockIdx.x * 64;
  int j0 = blockIdx.y * 32;
  int tid = threadIdx.x;
  int tl = tid & 63, jl = tid >> 6;             // 64 t-lanes x 4 j per pass
#pragma unroll
  for (int pass = 0; pass < 8; ++pass) {
    int j = j0 + pass * 4 + jl;
    int c = j >> 6, fq = j & 63;
    const float* base = x + (((long)(b * C + c) * F + 4 * fq) * (long)T + t0 + tl);
    float v = fmaxf(fmaxf(base[0], base[(long)T]),
                    fmaxf(base[2L * T], base[3L * T]));
    lds[pass * 4 + jl][tl] = (_Float16)v;
  }
  __syncthreads();
  int jl2 = tid & 31, bt = tid >> 5;            // 32 j-lanes x 8 t per pass
#pragma unroll
  for (int pass = 0; pass < 8; ++pass) {
    int tl2 = pass * 8 + bt;
    o[((long)b * T + t0 + tl2) * IN_DIM + j0 + jl2] = lds[jl2][tl2];
  }
}

// ---------------------------------------------------------------------------
// Kernel 2: q = Xm*Wq^T + bq ; k = Xf*Wk^T + bk   (WMMA f16 -> f32 -> f16)
// grid (BT/64, 2), block 128 (4 waves, one 16-row M tile each)
// ---------------------------------------------------------------------------
__global__ __launch_bounds__(128) void qk_gemm_kernel(
    const _Float16* __restrict__ pm, const _Float16* __restrict__ pf,
    const _Float16* __restrict__ wq, const _Float16* __restrict__ wk,
    const float* __restrict__ bq, const float* __restrict__ bk,
    _Float16* __restrict__ qo, _Float16* __restrict__ ko) {
  const _Float16* A;
  const _Float16* W;
  const float* bias;
  _Float16* out;
  if (blockIdx.y == 0) { A = pm; W = wq; bias = bq; out = qo; }
  else                 { A = pf; W = wk; bias = bk; out = ko; }

  int wave = threadIdx.x >> 5, lane = threadIdx.x & 31;
  long tile = (long)blockIdx.x * 4 + wave;      // 0..2047
  long row0 = tile * 16;
  int m = lane & 15, hi = lane >> 4;

  v8f acc[4];
#pragma unroll
  for (int nt = 0; nt < 4; ++nt)
#pragma unroll
    for (int r = 0; r < 8; ++r) acc[nt][r] = 0.f;

#pragma unroll
  for (int kk = 0; kk < 4; ++kk) {              // K = 128 in steps of 32
    // A fragment (ISA 16-bit A layout): element i -> K = kk*32 + i + 8*hi (+8 if i>=8)
    const _Float16* arow = A + (row0 + m) * IN_DIM + kk * 32 + 8 * hi;
    v16h a = join16(*(const v8h*)arow, *(const v8h*)(arow + 16));
#pragma unroll
    for (int nt = 0; nt < 4; ++nt) {
      // B fragment: element i -> (K = kk*32 + i + 8*hi (+8 if i>=8), N = lane&15)
      int p = nt * 16 + m;
      const _Float16* wrow = W + p * IN_DIM + kk * 32 + 8 * hi;
      v16h bf = join16(*(const v8h*)wrow, *(const v8h*)(wrow + 16));
      acc[nt] = __builtin_amdgcn_wmma_f32_16x16x32_f16(
          false, a, false, bf, (short)0, acc[nt], false, false);
    }
  }

#pragma unroll
  for (int nt = 0; nt < 4; ++nt) {
    int p = nt * 16 + m;
    float bv = bias[p];
#pragma unroll
    for (int r = 0; r < 8; ++r) {
      int mm = r + 8 * hi;                      // C/D layout: VGPR r -> M = r + 8*(lane>=16)
      out[(row0 + mm) * P + p] = (_Float16)(acc[nt][r] + bv);
    }
  }
}

// ---------------------------------------------------------------------------
// Kernel 3: banded Q.K^T (3 WMMA s-tiles per 16-row t-tile) + mask + softmax
// grid (BT/64), block 128 (4 waves)
// ---------------------------------------------------------------------------
__global__ __launch_bounds__(128) void corr_kernel(const _Float16* __restrict__ q,
                                                   const _Float16* __restrict__ k,
                                                   float* __restrict__ corr_out) {
  __shared__ float sc[4][16][49];               // per-wave 16 x 48 (+1 pad)
  int wave = threadIdx.x >> 5, lane = threadIdx.x & 31;
  long tile = (long)blockIdx.x * 4 + wave;
  long t0 = tile * 16;
  long batch_base = (t0 / T) * (long)T;
  int m = lane & 15, hi = lane >> 4;

  v8f acc[3];
#pragma unroll
  for (int j = 0; j < 3; ++j)
#pragma unroll
    for (int r = 0; r < 8; ++r) acc[j][r] = 0.f;

#pragma unroll
  for (int kk = 0; kk < 2; ++kk) {              // P = 64 in steps of 32
    const _Float16* qrow = q + (t0 + m) * P + kk * 32 + 8 * hi;
    v16h a = join16(*(const v8h*)qrow, *(const v8h*)(qrow + 16));
#pragma unroll
    for (int j = 0; j < 3; ++j) {
      long s = t0 - 32 + j * 16 + m;            // K row for N = lane&15
      s = (s < batch_base) ? batch_base : s;    // clamp (masked later)
      const _Float16* krow = k + s * P + kk * 32 + 8 * hi;
      v16h bf = join16(*(const v8h*)krow, *(const v8h*)(krow + 16));
      acc[j] = __builtin_amdgcn_wmma_f32_16x16x32_f16(
          false, a, false, bf, (short)0, acc[j], false, false);
    }
  }

#pragma unroll
  for (int j = 0; j < 3; ++j)
#pragma unroll
    for (int r = 0; r < 8; ++r)
      sc[wave][r + 8 * hi][j * 16 + m] = acc[j][r];
  __syncthreads();

  if (lane < 16) {
    int mm = lane;
    long t = t0 + mm;
    int t_local = (int)(t - batch_base);
    float v[D];
    float mx = -3.0e38f;
#pragma unroll
    for (int d = 0; d < D; ++d) {
      // col c = (t + d - 31) - (t0 - 32) = mm + 1 + d
      float val = sc[wave][mm][mm + 1 + d];
      val = (t_local + d >= D - 1) ? val : -1e30f;
      v[d] = val;
      mx = fmaxf(mx, val);
    }
    float sum = 0.f;
#pragma unroll
    for (int d = 0; d < D; ++d) { float e = __expf(v[d] - mx); v[d] = e; sum += e; }
    float inv = 1.f / sum;
    float* cr = corr_out + t * D;
#pragma unroll
    for (int d = 0; d < D; ++d) cr[d] = v[d] * inv;
  }
}

// ---------------------------------------------------------------------------
// Kernel 4: out[b,c,f,t] = sum_d corr[b,t,d] * xf[b,c,f, t+d-31]
// grid (T/256, B*C*F), block 256
// ---------------------------------------------------------------------------
__global__ __launch_bounds__(256) void out_win_kernel(const float* __restrict__ xf,
                                                      const float* __restrict__ corr,
                                                      float* __restrict__ out) {
  __shared__ float tile[256 + D - 1];
  long row = blockIdx.y;                        // b*C*F + c*F + f
  int b = (int)(row / (C * F));
  long t0 = (long)blockIdx.x * 256;
  int tid = threadIdx.x;
  const float* xrow = xf + row * (long)T;
  for (int i = tid; i < 256 + D - 1; i += 256) {
    long gt = t0 - (D - 1) + i;
    tile[i] = (gt >= 0) ? xrow[gt] : 0.f;
  }
  __syncthreads();
  long t = t0 + tid;
  const float4* c4 = (const float4*)(corr + ((long)b * T + t) * D);
  float acc = 0.f;
#pragma unroll
  for (int dq = 0; dq < 8; ++dq) {
    float4 cv = c4[dq];
    int base = tid + dq * 4;
    acc += cv.x * tile[base]     + cv.y * tile[base + 1] +
           cv.z * tile[base + 2] + cv.w * tile[base + 3];
  }
  out[row * (long)T + t] = acc;
}

// ---------------------------------------------------------------------------
extern "C" void kernel_launch(void* const* d_in, const int* in_sizes, int n_in,
                              void* d_out, int out_size, void* d_ws, size_t ws_size,
                              hipStream_t stream) {
  const float* xm = (const float*)d_in[0];
  const float* xf = (const float*)d_in[1];
  const float* Wq = (const float*)d_in[2];
  const float* bq = (const float*)d_in[3];
  const float* Wk = (const float*)d_in[4];
  const float* bk = (const float*)d_in[5];

  float* out = (float*)d_out;
  float* corr = out + OUT_ELEMS;

  _Float16* ws  = (_Float16*)d_ws;              // ~25.2 MB of workspace used
  _Float16* pm  = ws;
  _Float16* pf  = pm + (size_t)BT * IN_DIM;
  _Float16* q16 = pf + (size_t)BT * IN_DIM;
  _Float16* k16 = q16 + (size_t)BT * P;
  _Float16* wq16 = k16 + (size_t)BT * P;
  _Float16* wk16 = wq16 + (size_t)P * IN_DIM;

  cast_w_kernel<<<32, 256, 0, stream>>>(Wq, Wk, wq16, wk16);
  pool_kernel<<<dim3(T / 64, IN_DIM / 32, 2 * B), 256, 0, stream>>>(xm, xf, pm, pf);
  qk_gemm_kernel<<<dim3(BT / 64, 2), 128, 0, stream>>>(pm, pf, wq16, wk16, bq, bk, q16, k16);
  corr_kernel<<<BT / 64, 128, 0, stream>>>(q16, k16, corr);
  out_win_kernel<<<dim3(T / 256, B * C * F), 256, 0, stream>>>(xf, corr, out);
}
